// MLPWithV_60404420051341
// MI455X (gfx1250) — 16-bit and 8-bit matrix operand order repaired and checked by reading, compiled, not run on hardware
//
#include <hip/hip_runtime.h>
#include <stdint.h>


typedef __attribute__((ext_vector_type(2)))  int            i32x2;
typedef __attribute__((ext_vector_type(4)))  int            i32x4;
typedef __attribute__((ext_vector_type(8)))  int            i32x8;
typedef __attribute__((ext_vector_type(8)))  float          f32x8;
typedef __attribute__((ext_vector_type(8)))  unsigned short u16x8;
typedef __attribute__((ext_vector_type(16))) unsigned short u16x16;
typedef __attribute__((ext_vector_type(16))) __bf16         bf16x16;

#define ROWS     32
#define THREADS  64
#define KC       256
#define KQ_STEPS (KC / 64)
#define KU_STEPS (KC / 32)
#define ACTW     256

struct LayerP {
  const int*            iq;
  const int*            iuq;
  const float*          sw;
  const float*          bias;
  const unsigned short* upack;
  const signed char*    qpack;
  int nq, nuq, kpad, fo;
};

struct Params {
  LayerP       L[5];
  const float* x;
  float*       out;
  int          batch;
  int          inw;
};

__device__ __forceinline__ unsigned short f2bf(float f) {
  unsigned int u = __float_as_uint(f);
  u += 0x7FFFu + ((u >> 16) & 1u);
  return (unsigned short)(u >> 16);
}


__global__ void pack_qw_i8(const int* __restrict__ qw, signed char* __restrict__ dst,
                           int fo, int nq, int kpad) {
  int i = blockIdx.x * blockDim.x + threadIdx.x;
  int tot = fo * kpad;
  if (i >= tot) return;
  int n = i / kpad, k = i - n * kpad;
  dst[i] = (k < nq) ? (signed char)qw[n * nq + k] : (signed char)0;
}

__global__ void pack_uqw_bf16(const float* __restrict__ w, unsigned short* __restrict__ dst,
                              int fo, int nuq, int kpad) {
  int i = blockIdx.x * blockDim.x + threadIdx.x;
  int tot = fo * kpad;
  if (i >= tot) return;
  int n = i / kpad, k = i - n * kpad;
  dst[i] = (k < nuq) ? f2bf(w[n * nuq + k]) : (unsigned short)0;
}


template <int NT, bool FROM_GLOBAL, bool RELU>
__device__ __forceinline__ void do_layer(
    const LayerP lp, const float* __restrict__ x, float* s_act,
    unsigned short* s_u, signed char* s_q,
    float* s_sx, float* s_qs, float* s_red,
    long rowbase, int inw, int batch,
    int tid, int lg, int ln, int m0) {

  {
    int r = tid >> 1, part = tid & 1;
    const long grow   = (rowbase + r) * (long)inw;
    const bool gvalid = (rowbase + r) < batch;
    float ma = 0.0f, mb = 0.0f, mc = 0.0f, md = 0.0f;
    int j = part;
    for (; j + 6 < lp.nq; j += 8) {
      int c0 = lp.iq[j], c1 = lp.iq[j + 2], c2 = lp.iq[j + 4], c3 = lp.iq[j + 6];
      float v0, v1, v2, v3;
      if (FROM_GLOBAL) {
        v0 = gvalid ? x[grow + c0] : 0.0f;
        v1 = gvalid ? x[grow + c1] : 0.0f;
        v2 = gvalid ? x[grow + c2] : 0.0f;
        v3 = gvalid ? x[grow + c3] : 0.0f;
      } else {
        v0 = s_act[r * ACTW + c0]; v1 = s_act[r * ACTW + c1];
        v2 = s_act[r * ACTW + c2]; v3 = s_act[r * ACTW + c3];
      }
      ma = fmaxf(ma, fabsf(v0)); mb = fmaxf(mb, fabsf(v1));
      mc = fmaxf(mc, fabsf(v2)); md = fmaxf(md, fabsf(v3));
    }
    for (; j < lp.nq; j += 2) {
      int   c = lp.iq[j];
      float v = FROM_GLOBAL ? (gvalid ? x[grow + c] : 0.0f)
                            : s_act[r * ACTW + c];
      ma = fmaxf(ma, fabsf(v));
    }
    s_red[tid] = fmaxf(fmaxf(ma, mb), fmaxf(mc, md));
  }
  __syncthreads();
  if (tid < ROWS) {
    float sx = fmaxf(fmaxf(s_red[2 * tid], s_red[2 * tid + 1]), 1e-8f);
    s_sx[tid] = sx;
    s_qs[tid] = 127.0f / sx;
  }
  __syncthreads();

  int offl[NT];
#pragma unroll
  for (int nt = 0; nt < NT; ++nt) {
    const int n   = nt * 16 + ln;
    const int ncl = (n < lp.fo) ? n : (lp.fo - 1);
    offl[nt] = ncl * lp.kpad + 16 * lg;
  }

  f32x8 facc[NT] = {};
  i32x8 iacc[NT] = {};

  for (int kb = 0; kb < lp.kpad; kb += KC) {
    for (int i = tid; i < ROWS * KC; i += THREADS) {
      const int  r  = i >> 8;
      const int  k  = i & (KC - 1);
      const int  kq = kb + k;
      const long grow   = (rowbase + r) * (long)inw;
      const bool gvalid = (rowbase + r) < batch;
      signed char q = 0;
      if (kq < lp.nq) {
        int   c = lp.iq[kq];
        float v = FROM_GLOBAL ? (gvalid ? x[grow + c] : 0.0f)
                              : s_act[r * ACTW + c];
        q = (signed char)(int)__builtin_rintf(v * s_qs[r]);
      }
      s_q[i] = q;
      unsigned short u = 0;
      if (kq < lp.nuq) {
        int   c = lp.iuq[kq];
        float v = FROM_GLOBAL ? (gvalid ? x[grow + c] : 0.0f)
                              : s_act[r * ACTW + c];
        u = f2bf(v);
      }
      s_u[i] = u;
    }
    __syncthreads();

    const int arow = (m0 + ln) * KC;
    i32x8   aq[KQ_STEPS];
    bf16x16 au[KU_STEPS];
#pragma unroll
    for (int t = 0; t < KQ_STEPS; ++t) {
      const int kk = t * 64;
      i32x2 g0 = *(const i32x2*)(s_q + arow + kk +      16 * lg);
      i32x2 g1 = *(const i32x2*)(s_q + arow + kk + 8  + 16 * lg);
      i32x2 g2 = *(const i32x2*)(s_q + arow + kk + 32 + 16 * lg);
      i32x2 g3 = *(const i32x2*)(s_q + arow + kk + 40 + 16 * lg);
      i32x8 a;
      a[0] = g0[0]; a[1] = g0[1]; a[2] = g1[0]; a[3] = g1[1];
      a[4] = g2[0]; a[5] = g2[1]; a[6] = g3[0]; a[7] = g3[1];
      aq[t] = a;
    }
#pragma unroll
    for (int t = 0; t < KU_STEPS; ++t) {
      const int kk = t * 32;
      u16x8  al = *(const u16x8*)(s_u + arow + kk + 16 * lg);
      u16x8  ah = *(const u16x8*)(s_u + arow + kk + 16 * lg + 8);
      u16x16 ar = __builtin_shufflevector(al, ah,
          0, 1, 2, 3, 4, 5, 6, 7, 8, 9, 10, 11, 12, 13, 14, 15);
      au[t] = __builtin_bit_cast(bf16x16, ar);
    }

    const signed char*    qbase = lp.qpack + kb;
    const unsigned short* ubase = lp.upack + kb;
#pragma unroll
    for (int nt = 0; nt < NT; ++nt) {
      const int o = offl[nt];

#pragma unroll
      for (int t = 0; t < KQ_STEPS; ++t) {
        i32x4 bl = *(const i32x4*)(qbase + o + t * 64);
        i32x4 bh = *(const i32x4*)(qbase + o + t * 64 + 32);
        i32x8 b  = __builtin_shufflevector(bl, bh, 0, 1, 2, 3, 4, 5, 6, 7);
        iacc[nt] = __builtin_amdgcn_wmma_i32_16x16x64_iu8(
            true, aq[t], true, b, iacc[nt], false, false);
      }

#pragma unroll
      for (int t = 0; t < KU_STEPS; ++t) {
        u16x8  bl = *(const u16x8*)(ubase + o + t * 32);
        u16x8  bh = *(const u16x8*)(ubase + o + t * 32 + 8);
        u16x16 br = __builtin_shufflevector(bl, bh,
            0, 1, 2, 3, 4, 5, 6, 7, 8, 9, 10, 11, 12, 13, 14, 15);
        bf16x16 b = __builtin_bit_cast(bf16x16, br);
        facc[nt] = __builtin_amdgcn_wmma_f32_16x16x32_bf16(
            false, au[t], false, b, (short)0, facc[nt], false, false);
      }
    }
    __syncthreads();
  }

#pragma unroll
  for (int nt = 0; nt < NT; ++nt) {
    int   n   = nt * 16 + ln;
    bool  nv  = n < lp.fo;
    float swn = nv ? lp.sw[n]   : 0.0f;
    float bn  = nv ? lp.bias[n] : 0.0f;
#pragma unroll
    for (int v = 0; v < 8; ++v) {
      int   m  = m0 + v + 8 * lg;
      float sx = s_sx[m];
      float y  = facc[nt][v]
               + (float)iacc[nt][v] * (sx * swn * (1.0f / 16129.0f))
               + bn;
      if (RELU) y = fmaxf(y, 0.0f);
      if (nv) s_act[m * ACTW + n] = y;
    }
  }
  __syncthreads();
}


__global__ __launch_bounds__(THREADS) void fused_mlp(Params p) {
  __shared__ __align__(16) float          s_act[ROWS * ACTW];
  __shared__ __align__(16) unsigned short s_u[ROWS * KC];
  __shared__ __align__(16) signed char    s_q[ROWS * KC];
  __shared__               float          s_sx[ROWS];
  __shared__               float          s_qs[ROWS];
  __shared__               float          s_red[THREADS];

  const int  tid  = threadIdx.x;
  const int  wv   = tid >> 5;
  const int  lane = tid & 31;
  const int  lg   = lane >> 4;
  const int  ln   = lane & 15;
  const int  m0   = wv * 16;
  const long rowbase = (long)blockIdx.x * ROWS;

  do_layer<16, true,  true >(p.L[0], p.x, s_act, s_u, s_q, s_sx, s_qs, s_red,
                             rowbase, p.inw, p.batch, tid, lg, ln, m0);
  do_layer<4,  false, true >(p.L[1], p.x, s_act, s_u, s_q, s_sx, s_qs, s_red,
                             rowbase, p.inw, p.batch, tid, lg, ln, m0);
  do_layer<1,  false, true >(p.L[2], p.x, s_act, s_u, s_q, s_sx, s_qs, s_red,
                             rowbase, p.inw, p.batch, tid, lg, ln, m0);
  do_layer<1,  false, true >(p.L[3], p.x, s_act, s_u, s_q, s_sx, s_qs, s_red,
                             rowbase, p.inw, p.batch, tid, lg, ln, m0);
  do_layer<1,  false, false>(p.L[4], p.x, s_act, s_u, s_q, s_sx, s_qs, s_red,
                             rowbase, p.inw, p.batch, tid, lg, ln, m0);

  const int FOUT = p.L[4].fo;
  for (int i = tid; i < ROWS * FOUT; i += THREADS) {
    int  r  = i / FOUT, c = i - r * FOUT;
    long gr = rowbase + r;
    if (gr < p.batch) p.out[gr * (long)FOUT + c] = s_act[r * ACTW + c];
  }
}


extern "C" void kernel_launch(void* const* d_in, const int* in_sizes, int n_in,
                              void* d_out, int out_size, void* d_ws, size_t ws_size,
                              hipStream_t stream) {
  (void)ws_size;
  Params P;
  unsigned char* ws  = (unsigned char*)d_ws;
  size_t         off = 0;
  auto align256 = [](size_t o) { return (o + 255) & ~(size_t)255; };

  for (int l = 0; l < 5; ++l) {
    const float* uqw = (const float*)d_in[6 * l + 0];
    const int*   qw  = (const int*)  d_in[6 * l + 1];
    const float* sw  = (const float*)d_in[6 * l + 2];
    const float* b   = (const float*)d_in[6 * l + 3];
    const int*   iq  = (const int*)  d_in[6 * l + 4];
    const int*   iuq = (const int*)  d_in[6 * l + 5];
    int fo   = in_sizes[6 * l + 2];
    int nq   = in_sizes[6 * l + 4];
    int nuq  = in_sizes[6 * l + 5];
    int kmax = nq > nuq ? nq : nuq;
    int kpad = (kmax + KC - 1) & ~(KC - 1);

    off = align256(off);
    signed char* qpack = (signed char*)(ws + off);
    off += (size_t)fo * kpad;
    off = align256(off);
    unsigned short* upack = (unsigned short*)(ws + off);
    off += (size_t)fo * kpad * 2;

    int tq = fo * kpad;
    pack_qw_i8<<<(tq + 255) / 256, 256, 0, stream>>>(qw, qpack, fo, nq, kpad);
    pack_uqw_bf16<<<(tq + 255) / 256, 256, 0, stream>>>(uqw, upack, fo, nuq, kpad);

    P.L[l].iq = iq;   P.L[l].iuq = iuq;
    P.L[l].sw = sw;   P.L[l].bias = b;
    P.L[l].upack = upack; P.L[l].qpack = qpack;
    P.L[l].nq = nq;   P.L[l].nuq = nuq;
    P.L[l].kpad = kpad;
    P.L[l].fo = fo;
  }

  const int fo5   = P.L[4].fo;
  const int batch = out_size / fo5;
  const long xel  = (long)in_sizes[n_in - 1];
  P.x     = (const float*)d_in[n_in - 1];
  P.out   = (float*)d_out;
  P.batch = batch;
  P.inw   = (int)(xel / batch);

  int nblocks = (batch + ROWS - 1) / ROWS;
  fused_mlp<<<nblocks, THREADS, 0, stream>>>(P);
}
